// LSTMRegressor_15693810500384
// MI455X (gfx1250) — matrix operand order repaired and checked by reading, not compiled, not run on hardware
//
#include <hip/hip_runtime.h>
#include <stdint.h>

typedef __attribute__((ext_vector_type(16))) _Float16 v16h;
typedef __attribute__((ext_vector_type(8)))  _Float16 v8h;
typedef __attribute__((ext_vector_type(8)))  float    v8f;

#define TSEQ   1024
#define HID    64
#define CHUNK  64
#define NCHUNK (TSEQ / CHUNK)
#define ROWS   16          // batch rows per workgroup
#define HPITCH 72          // halves per h row (144B, padded: 36 dwords -> conflict-free)
#define XROWF  (CHUNK * 3) // floats per x row per chunk (192 floats = 768B)

__device__ __forceinline__ float fast_tanh(float x) {
#if __has_builtin(__builtin_amdgcn_tanhf)
  return __builtin_amdgcn_tanhf(x);
#else
  // tanh(x) = 2/(1+exp(-2x)) - 1 ; exp(-2x) = exp2(-2*log2(e)*x)
  float e = __builtin_amdgcn_exp2f(-2.885390082f * x);
  return __builtin_amdgcn_rcpf(1.0f + e) * 2.0f - 1.0f;
#endif
}
__device__ __forceinline__ float fast_sigmoid(float x) {
  return 0.5f * fast_tanh(0.5f * x) + 0.5f;
}

__device__ __forceinline__ v8f wmma_f16(v16h a, v16h b, v8f c) {
  // (neg_a, A, neg_b, B, c_mod, C, reuse_a, reuse_b)
  return __builtin_amdgcn_wmma_f32_16x16x32_f16(false, a, false, b, (short)0, c,
                                                false, false);
}

// Load one 16x32 f16 A-tile chunk from an LDS row (row-major, 64 halves of h).
// ISA A layout: lanes 0-15 hold K = {0..7,16..23}; lanes 16-31 hold K = {8..15,24..31}.
__device__ __forceinline__ v16h load_hA(const _Float16* rowBase, int hl) {
  const v8h lo = *(const v8h*)(rowBase + hl * 8);        // K = 8*hl .. +7
  const v8h hi = *(const v8h*)(rowBase + 16 + hl * 8);   // K = 16 + 8*hl .. +7
  return __builtin_shufflevector(lo, hi, 0, 1, 2, 3, 4, 5, 6, 7,
                                 8, 9, 10, 11, 12, 13, 14, 15);
}

__global__ __launch_bounds__(128) void lstm_regressor_kernel(
    const float* __restrict__ x,
    const float* __restrict__ W_ih, const float* __restrict__ W_hh,
    const float* __restrict__ b_ih, const float* __restrict__ b_hh,
    const float* __restrict__ W_sbp, const float* __restrict__ b_sbp,
    const float* __restrict__ W_dbp, const float* __restrict__ b_dbp,
    float* __restrict__ out, int B) {
  __shared__ __align__(16) float    xsh[2][ROWS][XROWF];   // 24 KB, double-buffered x chunks
  __shared__ __align__(16) _Float16 hsh[2][ROWS][HPITCH];  // 4.5 KB, double-buffered h

  const int tid  = threadIdx.x;
  const int wave = tid >> 5;
  const int lane = tid & 31;
  const int li   = lane & 15;   // N column / M row selector
  const int hl   = lane >> 4;   // lane half
  const int b0   = blockIdx.x * ROWS;
  const int jcol = wave * 16 + li;  // this lane's hidden unit within [0,64)

  const float* xg = x + (size_t)b0 * TSEQ * 3;  // wave-uniform global base

  // ---- async prefetch of one 64-timestep x chunk into LDS (6 insts/wave) ----
  auto prefetch = [&](int chunk) {
    const int buf = chunk & 1;
    const unsigned ldsBase = (unsigned)(size_t)&xsh[buf][0][0];
#pragma unroll
    for (int g = 0; g < 2; ++g) {
      const int row = wave * 4 + g * 2 + hl;  // 16 rows over 4 waves
#pragma unroll
      for (int k = 0; k < 3; ++k) {
        const unsigned off = (unsigned)(k * 256 + li * 16);
        const unsigned lds = ldsBase + (unsigned)(row * (XROWF * 4)) + off;
        const unsigned gof = (unsigned)(row * (TSEQ * 3 * 4) + chunk * (CHUNK * 3 * 4)) + off;
        asm volatile("global_load_async_to_lds_b128 %0, %1, %2 offset:0"
                     :: "v"(lds), "v"(gof), "s"(xg)
                     : "memory");
      }
    }
  };

  prefetch(0);  // overlap with weight setup below

  // ---- zero h double-buffer (h0 = 0) ----
  {
    _Float16* hz = &hsh[0][0][0];
    for (int i = tid; i < 2 * ROWS * HPITCH; i += 128) hz[i] = (_Float16)0.f;
  }

  // ---- resident B operands: W_hh (8 tiles) and padded W_ih (4 tiles), f16 ----
  // B layout (32x16): lanes 0-15 hold K={0..7,16..23}, lanes 16-31 hold K={8..15,24..31}.
  v16h Bih[4];
  v16h Bhh[4][2];
  v8f  cbias[4];
#pragma unroll
  for (int g = 0; g < 4; ++g) {
    const int j = g * HID + jcol;  // gate-row of W_ih / W_hh
    v16h bi;
#pragma unroll
    for (int t = 0; t < 16; ++t) bi[t] = (_Float16)0.f;
    if (hl == 0) {  // only K=0..2 nonzero, and only in lanes 0-15
      bi[0] = (_Float16)W_ih[j * 3 + 0];
      bi[1] = (_Float16)W_ih[j * 3 + 1];
      bi[2] = (_Float16)W_ih[j * 3 + 2];
    }
    Bih[g] = bi;
#pragma unroll
    for (int kc = 0; kc < 2; ++kc) {
      const float4* wr = (const float4*)(W_hh + (size_t)j * HID + kc * 32 + hl * 8);
      v16h bh;
#pragma unroll
      for (int q = 0; q < 4; ++q) {
        float4 f = wr[q + (q >> 1) * 2];
        bh[q * 4 + 0] = (_Float16)f.x;
        bh[q * 4 + 1] = (_Float16)f.y;
        bh[q * 4 + 2] = (_Float16)f.z;
        bh[q * 4 + 3] = (_Float16)f.w;
      }
      Bhh[g][kc] = bh;
    }
    const float bias = b_ih[j] + b_hh[j];
#pragma unroll
    for (int e = 0; e < 8; ++e) cbias[g][e] = bias;
  }

  float cst[8];
#pragma unroll
  for (int e = 0; e < 8; ++e) cst[e] = 0.f;

  int rd = 0;  // h read buffer
#pragma unroll 1
  for (int c = 0; c < NCHUNK; ++c) {
    if (c + 1 < NCHUNK) {
      prefetch(c + 1);
      asm volatile("s_wait_asynccnt 0x6" ::: "memory");  // chunk c done (in-order)
    } else {
      asm volatile("s_wait_asynccnt 0x0" ::: "memory");
    }
    __syncthreads();  // all waves' portions of chunk c visible
    const int xb = c & 1;

#pragma unroll 1
    for (int tl = 0; tl < CHUNK; ++tl) {
      // x A-tile: rows m=0..15 in lanes 0-15, halves K=0..2; all else zero
      v16h xa;
#pragma unroll
      for (int t = 0; t < 16; ++t) xa[t] = (_Float16)0.f;
      if (hl == 0) {
        const float* xr = &xsh[xb][li][tl * 3];
        xa[0] = (_Float16)xr[0];
        xa[1] = (_Float16)xr[1];
        xa[2] = (_Float16)xr[2];
      }
      // h A-tiles (K = 0..31 and 32..63) from LDS
      const _Float16* rowBase = &hsh[rd][li][0];
      const v16h hA0 = load_hA(rowBase, hl);
      const v16h hA1 = load_hA(rowBase + 32, hl);

      // gates = bias + x@W_ihT + h@W_hhT  (12 WMMAs)
      v8f acc[4];
#pragma unroll
      for (int g = 0; g < 4; ++g) {
        v8f a = cbias[g];
        a = wmma_f16(xa, Bih[g], a);
        a = wmma_f16(hA0, Bhh[g][0], a);
        a = wmma_f16(hA1, Bhh[g][1], a);
        acc[g] = a;
      }

      const int wr = rd ^ 1;
#pragma unroll
      for (int e = 0; e < 8; ++e) {  // rows m = e + 8*hl, column jcol
        const float iv = fast_sigmoid(acc[0][e]);
        const float fv = fast_sigmoid(acc[1][e]);
        const float gv = fast_tanh(acc[2][e]);
        const float ov = fast_sigmoid(acc[3][e]);
        cst[e] = fv * cst[e] + iv * gv;
        const float hv = ov * fast_tanh(cst[e]);
        hsh[wr][e + 8 * hl][jcol] = (_Float16)hv;
      }
      __syncthreads();  // h_t visible to all waves before next step
      rd = wr;
    }
  }

  // ---- final projections: sbp/dbp from h_T (16 rows, one lane each) ----
  if (wave == 0 && lane < 16) {
    const int m = lane;
    float s1 = 0.f, s2 = 0.f;
#pragma unroll
    for (int k = 0; k < HID; ++k) {
      const float hv = (float)hsh[rd][m][k];
      s1 += hv * W_sbp[k];
      s2 += hv * W_dbp[k];
    }
    out[b0 + m]     = s1 + b_sbp[0];
    out[B + b0 + m] = s2 + b_dbp[0];
  }
}

extern "C" void kernel_launch(void* const* d_in, const int* in_sizes, int n_in,
                              void* d_out, int out_size, void* d_ws, size_t ws_size,
                              hipStream_t stream) {
  const float* x     = (const float*)d_in[0];
  const float* W_ih  = (const float*)d_in[1];
  const float* W_hh  = (const float*)d_in[2];
  const float* b_ih  = (const float*)d_in[3];
  const float* b_hh  = (const float*)d_in[4];
  const float* W_sbp = (const float*)d_in[5];
  const float* b_sbp = (const float*)d_in[6];
  const float* W_dbp = (const float*)d_in[7];
  const float* b_dbp = (const float*)d_in[8];

  const int B = in_sizes[0] / (TSEQ * 3);  // 4096
  dim3 grid(B / ROWS), block(128);
  lstm_regressor_kernel<<<grid, block, 0, stream>>>(
      x, W_ih, W_hh, b_ih, b_hh, W_sbp, b_sbp, W_dbp, b_dbp, (float*)d_out, B);
}
